// MultiLayerGCN_68298569941180
// MI455X (gfx1250) — compile-verified, matrix operand order checked by reading, not run on hardware
//
#include <hip/hip_runtime.h>

#define NN 4096
#define DD 256
#define THRESH 0.04f

#define MT 128      // block tile M (v)
#define NT 64       // block tile N (d)
#define KT 32       // K step (bf16 wmma K)
#define APITCH 40   // LDS row pitch in u16: 80B = 16B-aligned rows, 20-bank stagger

typedef __attribute__((ext_vector_type(16))) __bf16 v16bf;
typedef __attribute__((ext_vector_type(8)))  float  v8f;

// float -> bf16 (round to nearest even), stored as u16
static __device__ __forceinline__ unsigned short f2bf(float f) {
    unsigned u = __builtin_bit_cast(unsigned, f);
    unsigned r = u + 0x7FFFu + ((u >> 16) & 1u);
    return (unsigned short)(r >> 16);
}

// ---------------------------------------------------------------------------
// WMMA fragment loader from an LDS tile stored [row][k], pitch in u16 units.
// 16-bit A/B layout (ISA 7.12.2): lanes 0-15 hold rows 0-15 with K in
// {0..7,16..23}; lanes 16-31 same rows, K in {8..15,24..31}; each VGPR holds
// two consecutive K values -> 8 dword LDS reads per fragment.
// ---------------------------------------------------------------------------
static __device__ __forceinline__ v16bf load_frag(const unsigned short* tile,
                                                  int pitch, int lane) {
    const int row = lane & 15;
    const int hi  = (lane >> 4) & 1;
    const unsigned int* p = (const unsigned int*)(tile + row * pitch);
    union { unsigned int u[8]; v16bf v; } f;
#pragma unroll
    for (int r = 0; r < 4; ++r) f.u[r] = p[r + 4 * hi];       // K = 2r(+8*hi)
#pragma unroll
    for (int r = 4; r < 8; ++r) f.u[r] = p[r + 4 + 4 * hi];   // K = 2r+8(+8*hi)
    return f.v;
}

// ---------------------------------------------------------------------------
// Zero the degree accumulators (ws is poisoned by the harness).
// ---------------------------------------------------------------------------
__global__ __launch_bounds__(256) void zero_kernel(float* __restrict__ p, int n) {
    const int i = blockIdx.x * 256 + threadIdx.x;
    if (i < n) p[i] = 0.f;
}

// ---------------------------------------------------------------------------
// Single fused pass over A (the only full read of A in the pipeline):
//  - maskT[v][u] = bf16(A[u][v] >= T)  (transposed via LDS, coalesced both ways)
//  - in_sum[v]  += popcount of column  (f32 atomics; integer counts -> exact,
//    order-independent -> deterministic across graph replays)
//  - out_sum[u] += popcount of row
// ---------------------------------------------------------------------------
__global__ __launch_bounds__(256) void mask_deg_kernel(
    const float* __restrict__ A, unsigned short* __restrict__ maskT,
    float* __restrict__ out_sum, float* __restrict__ in_sum) {
    __shared__ unsigned short tile[64][65];
    const int t = threadIdx.x;
    const int ubase = blockIdx.x * 64;
    const int vbase = blockIdx.y * 64;
    const int c = t & 63;

    float colcnt = 0.f;                      // column v = vbase + c
#pragma unroll
    for (int j = 0; j < 16; ++j) {
        const int r = (t >> 6) + 4 * j;      // u offset
        const float a = A[(size_t)(ubase + r) * NN + vbase + c];
        const bool m = (a >= THRESH);
        tile[c][r] = m ? (unsigned short)0x3F80u : (unsigned short)0u;
        colcnt += m ? 1.f : 0.f;
    }
    atomicAdd(&in_sum[vbase + c], colcnt);
    __syncthreads();

    float rowcnt = 0.f;                      // row u = ubase + c
#pragma unroll
    for (int j = 0; j < 16; ++j) {
        const int r2 = (t >> 6) + 4 * j;     // v offset
        const unsigned short m = tile[r2][c];
        maskT[(size_t)(vbase + r2) * NN + ubase + c] = m;   // coalesced u16
        rowcnt += (m != 0) ? 1.f : 0.f;
    }
    atomicAdd(&out_sum[ubase + c], rowcnt);
}

// deg -> rsqrt(deg + 1)  (+1 self loop; result always >= 1 so clamp is implied)
__global__ __launch_bounds__(256) void finalize_deg_kernel(
    const float* __restrict__ out_sum, const float* __restrict__ in_sum,
    float* __restrict__ out_isq, float* __restrict__ in_isq) {
    const int i = blockIdx.x * 256 + threadIdx.x;
    out_isq[i] = rsqrtf(out_sum[i] + 1.0f);
    in_isq[i]  = rsqrtf(in_sum[i] + 1.0f);
}

// X[u,k] = bf16(features[u,k] * out_isqrt[u])
__global__ __launch_bounds__(256) void scale_cvt_kernel(
    const float* __restrict__ F, const float* __restrict__ out_isq,
    unsigned short* __restrict__ X) {
    const int i = blockIdx.x * 256 + threadIdx.x;
    const int u = i >> 8; // DD == 256
    X[i] = f2bf(F[i] * out_isq[u]);
}

// WT[d][k] = bf16(W[k][d])  (tiny: 256x256)
__global__ __launch_bounds__(256) void wt_kernel(const float* __restrict__ W,
                                                 unsigned short* __restrict__ WT) {
    const int i = blockIdx.x * 256 + threadIdx.x;
    const int d = i >> 8, k = i & 255;
    WT[(size_t)d * DD + k] = f2bf(W[(size_t)k * DD + d]);
}

// ---------------------------------------------------------------------------
// GEMM 1: hW = X @ W.  A = X bf16 [NN][DD]; B = WT bf16 [DD][DD] (k-contig).
// Double-buffered LDS, b128 staging.  Writes hWf (f32) and hWbT (bf16,
// transposed [d][u]) so GEMM 2's B staging is a straight copy.
// ---------------------------------------------------------------------------
__global__ __launch_bounds__(256) void gemm_xw_kernel(
    const unsigned short* __restrict__ X,
    const unsigned short* __restrict__ WT,
    float* __restrict__ hWf,
    unsigned short* __restrict__ hWbT) {
    __shared__ __align__(16) unsigned short As[2][MT * APITCH];
    __shared__ __align__(16) unsigned short Bs[2][NT * APITCH];
    const int t = threadIdx.x;
    const int lane = t & 31, wave = t >> 5;
    const int vbase = blockIdx.x * MT;
    const int dbase = blockIdx.y * NT;
    const int r = t >> 2, q = t & 3;     // 16B chunk coords (row, quad)

    auto ldA = [&](int kk, int rr) {
        return *(const uint4*)(X + (size_t)(vbase + rr) * DD + kk + q * 8);
    };
    auto ldB = [&](int kk) {
        return *(const uint4*)(WT + (size_t)(dbase + r) * DD + kk + q * 8);
    };
    auto stA = [&](int buf, int rr, uint4 val) {
        *(uint4*)(&As[buf][rr * APITCH + q * 8]) = val;
    };
    auto stB = [&](int buf, uint4 val) {
        *(uint4*)(&Bs[buf][r * APITCH + q * 8]) = val;
    };

    stA(0, r, ldA(0, r)); stA(0, r + 64, ldA(0, r + 64)); stB(0, ldB(0));
    __syncthreads();

    v8f acc[4];
#pragma unroll
    for (int nt = 0; nt < 4; ++nt) acc[nt] = (v8f){0.f,0.f,0.f,0.f,0.f,0.f,0.f,0.f};

    const int NKI = DD / KT;  // 8
    for (int ki = 0; ki < NKI; ++ki) {
        const int cur = ki & 1;
        uint4 na0, na1, nb;
        const bool more = (ki + 1) < NKI;
        if (more) {
            const int kk = (ki + 1) * KT;
            na0 = ldA(kk, r); na1 = ldA(kk, r + 64); nb = ldB(kk);
        }
        v16bf a = load_frag(&As[cur][wave * 16 * APITCH], APITCH, lane);
#pragma unroll
        for (int nt = 0; nt < 4; ++nt) {
            v16bf b = load_frag(&Bs[cur][nt * 16 * APITCH], APITCH, lane);
            acc[nt] = __builtin_amdgcn_wmma_f32_16x16x32_bf16(
                false, a, false, b, (short)0, acc[nt], false, false);
        }
        if (more) { stA(cur ^ 1, r, na0); stA(cur ^ 1, r + 64, na1); stB(cur ^ 1, nb); }
        __syncthreads();
    }

    const int col = lane & 15, rhi = (lane >> 4) & 1;
#pragma unroll
    for (int nt = 0; nt < 4; ++nt) {
        const int d = dbase + nt * 16 + col;
#pragma unroll
        for (int rr = 0; rr < 8; ++rr) {
            const int v = vbase + wave * 16 + rr + 8 * rhi;
            const float val = acc[nt][rr];
            hWf[(size_t)v * DD + d] = val;
            hWbT[(size_t)d * NN + v] = f2bf(val);
        }
    }
}

// ---------------------------------------------------------------------------
// GEMM 2: agg[v,d] = sum_u mask[u,v]*hW[u,d].  Both operands pre-transposed,
// k-contiguous in global -> pure b128 copy staging + WMMA.  Epilogue fuses
// self loop (exact f32), in_isqrt scale, bias, ReLU, and next-layer input.
// ---------------------------------------------------------------------------
__global__ __launch_bounds__(256) void gemm_agg_kernel(
    const unsigned short* __restrict__ maskT,  // [NN][NN] bf16, [v][u]
    const unsigned short* __restrict__ hWbT,   // [DD][NN] bf16, [d][u]
    const float* __restrict__ hWf,             // [NN][DD]
    const float* __restrict__ in_isq,          // [NN]
    const float* __restrict__ bias,            // [DD]
    const float* __restrict__ out_isq,         // [NN]
    float* __restrict__ hOut,                  // [NN][DD]
    unsigned short* __restrict__ Xnext,        // [NN][DD] bf16
    int write_next) {
    __shared__ __align__(16) unsigned short As[2][MT * APITCH];
    __shared__ __align__(16) unsigned short Bs[2][NT * APITCH];
    const int t = threadIdx.x;
    const int lane = t & 31, wave = t >> 5;
    const int vbase = blockIdx.x * MT;
    const int dbase = blockIdx.y * NT;
    const int r = t >> 2, q = t & 3;

    auto ldA = [&](int kk, int rr) {
        return *(const uint4*)(maskT + (size_t)(vbase + rr) * NN + kk + q * 8);
    };
    auto ldB = [&](int kk) {
        return *(const uint4*)(hWbT + (size_t)(dbase + r) * NN + kk + q * 8);
    };
    auto stA = [&](int buf, int rr, uint4 val) {
        *(uint4*)(&As[buf][rr * APITCH + q * 8]) = val;
    };
    auto stB = [&](int buf, uint4 val) {
        *(uint4*)(&Bs[buf][r * APITCH + q * 8]) = val;
    };

    stA(0, r, ldA(0, r)); stA(0, r + 64, ldA(0, r + 64)); stB(0, ldB(0));
    __syncthreads();

    v8f acc[4];
#pragma unroll
    for (int nt = 0; nt < 4; ++nt) acc[nt] = (v8f){0.f,0.f,0.f,0.f,0.f,0.f,0.f,0.f};

    const int NKI = NN / KT;  // 128
    for (int ki = 0; ki < NKI; ++ki) {
        const int cur = ki & 1;
        uint4 na0, na1, nb;
        const bool more = (ki + 1) < NKI;
        if (more) {
            const int kk = (ki + 1) * KT;
            na0 = ldA(kk, r); na1 = ldA(kk, r + 64); nb = ldB(kk);
        }
        if (ki + 2 < NKI) {  // prefetch two tiles ahead -> global_prefetch_b8
            const int kp = (ki + 2) * KT;
            __builtin_prefetch(maskT + (size_t)(vbase + (t >> 1)) * NN + kp, 0, 0);
            __builtin_prefetch(hWbT + (size_t)(dbase + (t >> 2)) * NN + kp, 0, 0);
        }
        v16bf a = load_frag(&As[cur][wave * 16 * APITCH], APITCH, lane);
#pragma unroll
        for (int nt = 0; nt < 4; ++nt) {
            v16bf b = load_frag(&Bs[cur][nt * 16 * APITCH], APITCH, lane);
            acc[nt] = __builtin_amdgcn_wmma_f32_16x16x32_bf16(
                false, a, false, b, (short)0, acc[nt], false, false);
        }
        if (more) { stA(cur ^ 1, r, na0); stA(cur ^ 1, r + 64, na1); stB(cur ^ 1, nb); }
        __syncthreads();
    }

    const int col = lane & 15, rhi = (lane >> 4) & 1;
#pragma unroll
    for (int nt = 0; nt < 4; ++nt) {
        const int d = dbase + nt * 16 + col;
#pragma unroll
        for (int rr = 0; rr < 8; ++rr) {
            const int v = vbase + wave * 16 + rr + 8 * rhi;
            float val = (acc[nt][rr] + hWf[(size_t)v * DD + d]) * in_isq[v] + bias[d];
            val = fmaxf(val, 0.f);
            hOut[(size_t)v * DD + d] = val;
            if (write_next)
                Xnext[(size_t)v * DD + d] = f2bf(val * out_isq[v]);
        }
    }
}

// ---------------------------------------------------------------------------
extern "C" void kernel_launch(void* const* d_in, const int* in_sizes, int n_in,
                              void* d_out, int out_size, void* d_ws, size_t ws_size,
                              hipStream_t stream) {
    const float* Ag   = (const float*)d_in[0];
    const float* feat = (const float*)d_in[1];
    const float* W1   = (const float*)d_in[2];
    const float* b1   = (const float*)d_in[3];
    const float* W2   = (const float*)d_in[4];
    const float* b2   = (const float*)d_in[5];
    float* out = (float*)d_out;

    // workspace layout (bytes)
    char* ws = (char*)d_ws;
    float*          out_sum = (float*)(ws);                       // 16 KB
    float*          in_sum  = (float*)(ws + (16 << 10));          // 16 KB
    float*          out_isq = (float*)(ws + (32 << 10));          // 16 KB
    float*          in_isq  = (float*)(ws + (48 << 10));          // 16 KB
    unsigned short* WT      = (unsigned short*)(ws + (64 << 10)); // 128 KB
    unsigned short* X1      = (unsigned short*)(ws + (256 << 10));// 2 MB
    float*          hWf     = (float*)(ws + ((size_t)4 << 20));   // 4 MB
    unsigned short* hWbT    = (unsigned short*)(ws + ((size_t)8 << 20));  // 2 MB
    unsigned short* X2      = (unsigned short*)(ws + ((size_t)10 << 20)); // 2 MB
    unsigned short* maskT   = (unsigned short*)(ws + ((size_t)12 << 20)); // 32 MB

    zero_kernel<<<32, 256, 0, stream>>>(out_sum, 2 * NN);  // out_sum & in_sum
    mask_deg_kernel<<<dim3(NN / 64, NN / 64), 256, 0, stream>>>(Ag, maskT,
                                                                out_sum, in_sum);
    finalize_deg_kernel<<<NN / 256, 256, 0, stream>>>(out_sum, in_sum,
                                                      out_isq, in_isq);
    scale_cvt_kernel<<<(NN * DD) / 256, 256, 0, stream>>>(feat, out_isq, X1);

    dim3 g(NN / MT, DD / NT);  // 32 x 4

    // layer 1
    wt_kernel<<<(DD * DD) / 256, 256, 0, stream>>>(W1, WT);
    gemm_xw_kernel<<<g, 256, 0, stream>>>(X1, WT, hWf, hWbT);
    gemm_agg_kernel<<<g, 256, 0, stream>>>(maskT, hWbT, hWf, in_isq, b1, out_isq,
                                           out, X2, 1);
    // layer 2
    wt_kernel<<<(DD * DD) / 256, 256, 0, stream>>>(W2, WT);
    gemm_xw_kernel<<<g, 256, 0, stream>>>(X2, WT, hWf, hWbT);
    gemm_agg_kernel<<<g, 256, 0, stream>>>(maskT, hWbT, hWf, in_isq, b2, out_isq,
                                           out + (size_t)NN * DD, X2, 0);
}
